// MyMultiHeadRelativeAttention_4526895530601
// MI455X (gfx1250) — matrix operand order repaired and checked by reading, not compiled, not run on hardware
//
#include <hip/hip_runtime.h>

#define DM     1024
#define NH     16
#define DEPTH  64
#define BATCH  4
#define SEQ    1024
#define MAXREL 128
#define NREL   257   // 2*MAXREL + 1
#define RPAD   272   // NREL rounded up to multiple of 16

typedef __attribute__((ext_vector_type(16))) __bf16 v16bf;
typedef __attribute__((ext_vector_type(8)))  __bf16 v8bf;
typedef __attribute__((ext_vector_type(4)))  __bf16 v4bf;
typedef __attribute__((ext_vector_type(8)))  float  v8f;

typedef __attribute__((ext_vector_type(4))) unsigned int u32x4;
typedef __attribute__((ext_vector_type(8))) int          i32x8;
typedef __attribute__((ext_vector_type(4))) int          i32x4;

// Toolchain detect: the TDM header ships only with the amdgpu-toolchain
// (clang-23 / therock-10.0), whose tensor_load builtin takes 6 args.
#if __has_include(<hip/amd_detail/amd_gfx1250_TDM.h>)
#define ATHENA_TDM_6ARG 1
#endif

static __device__ __forceinline__ v16bf cat16(v8bf lo, v8bf hi) {
  return __builtin_shufflevector(lo, hi, 0,1,2,3,4,5,6,7,8,9,10,11,12,13,14,15);
}
static __device__ __forceinline__ v8f wmma_bf16(v16bf a, v16bf b, v8f c) {
  // D = A(16x32 bf16) * B(32x16 bf16) + C(16x16 f32)
  return __builtin_amdgcn_wmma_f32_16x16x32_bf16(false, a, false, b, (short)0, c, false, false);
}

// -------------------------------------------------------------- TDM 2-D load
// Builds a Tensor DMA Descriptor (D#) per CDNA5 ISA ch.8 and issues
// tensor_load_to_lds. 2-byte elements; tile_d1 rows of tile_d0 elements,
// row stride stride0 elements in global; optional LDS row padding
// (pad codes per D#: interval 0=2DW..7=256DW, amount 0=1DW..127=128DW).
static __device__ __forceinline__ void tdm_load_2d_bf16(
    unsigned lds_byte_addr, const void* gptr,
    unsigned tile_d0, unsigned tile_d1,
    unsigned long long stride0,
    unsigned pad_interval_code, unsigned pad_amount_code) {
  unsigned long long ga = (unsigned long long)(size_t)gptr;
  u32x4 g0;
  g0.x = 1u;                                            // count=1 (valid user D#)
  g0.y = lds_byte_addr;                                 // lds_addr
  g0.z = (unsigned)(ga & 0xffffffffu);                  // global_addr[31:0]
  g0.w = (unsigned)((ga >> 32) & 0x01ffffffu) | (2u << 30);  // addr[56:32], type=2
  unsigned d0 = (1u << 16)                              // data_size = 2 bytes
              | (1u << 20)                              // pad_enable
              | (pad_interval_code << 22)
              | (pad_amount_code << 25);
  unsigned d1 = (tile_d0 & 0xffffu) << 16;              // tensor_dim0[15:0] (== tile_d0)
  unsigned d2 = ((tile_d0 >> 16) & 0xffffu)             // tensor_dim0[31:16]
              | ((tile_d1 & 0xffffu) << 16);            // tensor_dim1[15:0] (== tile_d1)
  unsigned d3 = ((tile_d1 >> 16) & 0xffffu)
              | ((tile_d0 & 0xffffu) << 16);            // tile_dim0
  unsigned d4 = (tile_d1 & 0xffffu);                    // tile_dim1; tile_dim2 = 0
  unsigned d5 = (unsigned)(stride0 & 0xffffffffull);    // tensor_dim0_stride[31:0]
  unsigned d6 = (unsigned)((stride0 >> 32) & 0xffffull);// stride[47:32]; dim1_stride lo = 0
  i32x8 g1 = { (int)d0, (int)d1, (int)d2, (int)d3, (int)d4, (int)d5, (int)d6, 0 };
  i32x4 gz = { 0, 0, 0, 0 };
#if defined(ATHENA_TDM_6ARG)
  i32x8 gz8 = { 0, 0, 0, 0, 0, 0, 0, 0 };
  __builtin_amdgcn_tensor_load_to_lds(g0, g1, gz, gz, gz8, 0);
#else
  __builtin_amdgcn_tensor_load_to_lds(g0, g1, gz, gz, 0);
#endif
}

// ---------------------------------------------------------------- convert
__global__ void cvt_f32_bf16(const float* __restrict__ in, __bf16* __restrict__ out, int n) {
  int i = blockIdx.x * blockDim.x + threadIdx.x;
  int stride = gridDim.x * blockDim.x;
  for (; i < n; i += stride) out[i] = (__bf16)in[i];
}

// ---------------------------------------------------------------- GEMM: C = A*W + bias
// A: [M,K] bf16 row-major, W: [K,N] bf16 row-major, bias: [N] f32.
// Block tile 128x128, 8 waves (4m x 2n), each wave 2x4 WMMA C-tiles.
// A tile staged by the Tensor Data Mover (padded-row descriptor), W tile
// staged manually with an LDS transpose.
template <bool OUTBF>
__global__ __launch_bounds__(256)
void wmma_gemm_bias(const __bf16* __restrict__ A, const __bf16* __restrict__ W,
                    const float* __restrict__ bias, void* __restrict__ outp,
                    int M, int N, int K) {
  __shared__ __bf16 As[128 * 40];   // [row][k0..31], padded stride 40 (TDM pad: 16DW+4DW)
  __shared__ __bf16 Bt[128 * 40];   // transposed: [n][k0..31], padded stride 40
  const int tid  = threadIdx.x;
  const int lane = tid & 31;
  const int wave = tid >> 5;
  const int half = lane >> 4;
  const int l16  = lane & 15;
  const int m0 = blockIdx.y * 128;
  const int n0 = blockIdx.x * 128;
  const int wm = (wave & 3) * 32;
  const int wn = (wave >> 2) * 64;
  const unsigned asAddr = (unsigned)(size_t)(void*)&As[0];

  v8f acc[2][4];
#pragma unroll
  for (int i = 0; i < 2; ++i)
#pragma unroll
    for (int j = 0; j < 4; ++j)
      acc[i][j] = (v8f){0.f,0.f,0.f,0.f,0.f,0.f,0.f,0.f};

  for (int k0 = 0; k0 < K; k0 += 32) {
    // stage A tile (128 rows x 32 bf16, row stride K) via TDM; LDS rows padded
    // 64B + 16B -> stride 40 elems. Uniform branch: TDM ignores EXEC.
    if (wave == 0) {
      tdm_load_2d_bf16(asAddr, &A[(size_t)m0 * K + k0],
                       /*tile_d0=*/32, /*tile_d1=*/128, /*stride0=*/(unsigned long long)K,
                       /*pad_interval(16DW)=*/3, /*pad_amount(4DW)=*/3);
      __builtin_amdgcn_s_wait_tensorcnt(0);
    }
    // stage W tile (32 x 128) transposed into Bt[n][k]
#pragma unroll
    for (int it = 0; it < 4; ++it) {
      int idx = (tid + it * 256) * 4;
      int kk = idx >> 7, nn = idx & 127;
      v4bf t = *(const v4bf*)&W[(size_t)(k0 + kk) * N + n0 + nn];
#pragma unroll
      for (int e = 0; e < 4; ++e) Bt[(nn + e) * 40 + kk] = t[e];
    }
    if (k0 + 32 < K) {   // emits global_prefetch_b8
      __builtin_prefetch(&A[(size_t)(m0 + (tid >> 1)) * K + k0 + 32], 0, 1);
      __builtin_prefetch(&W[(size_t)(k0 + 32 + (tid >> 3)) * N + n0], 0, 1);
    }
    __syncthreads();

    v16bf af[2], bfr[4];
#pragma unroll
    for (int mi = 0; mi < 2; ++mi) {
      const __bf16* p = &As[(wm + mi * 16 + l16) * 40];
      af[mi] = cat16(*(const v8bf*)&p[8 * half], *(const v8bf*)&p[16 + 8 * half]);
    }
#pragma unroll
    for (int ni = 0; ni < 4; ++ni) {
      const __bf16* p = &Bt[(wn + ni * 16 + l16) * 40];
      bfr[ni] = cat16(*(const v8bf*)&p[8 * half], *(const v8bf*)&p[16 + 8 * half]);
    }
#pragma unroll
    for (int mi = 0; mi < 2; ++mi)
#pragma unroll
      for (int ni = 0; ni < 4; ++ni)
        acc[mi][ni] = wmma_bf16(af[mi], bfr[ni], acc[mi][ni]);
    __syncthreads();
  }

#pragma unroll
  for (int mi = 0; mi < 2; ++mi)
#pragma unroll
    for (int ni = 0; ni < 4; ++ni) {
      int col = n0 + wn + ni * 16 + l16;
      float bv = bias[col];
#pragma unroll
      for (int v = 0; v < 8; ++v) {
        int row = m0 + wm + mi * 16 + v + 8 * half;
        float r = acc[mi][ni][v] + bv;
        if (OUTBF) ((__bf16*)outp)[(size_t)row * N + col] = (__bf16)r;
        else       ((float*)outp)[(size_t)row * N + col] = r;
      }
    }
}

// ---------------------------------------------------------------- Qrel = Q_head x rel_table^T
// Qrel[bh][q][r] = sum_d Q[b,q,h*64+d] * rel[r][d];  grid (BH, S/128), 8 waves x 16 q-rows.
__global__ __launch_bounds__(256)
void qrel_kernel(const __bf16* __restrict__ Qb, const __bf16* __restrict__ relb,
                 __bf16* __restrict__ Qrel) {
  const int tid = threadIdx.x, lane = tid & 31, wave = tid >> 5;
  const int half = lane >> 4, l16 = lane & 15;
  const int bh = blockIdx.x, b = bh >> 4, h = bh & 15;
  const int q0 = blockIdx.y * 128 + wave * 16;
  const int q  = q0 + l16;

  const __bf16* qrow = &Qb[((size_t)(b * SEQ) + q) * DM + h * DEPTH];
  v16bf af[2];
#pragma unroll
  for (int ks = 0; ks < 2; ++ks)
    af[ks] = cat16(*(const v8bf*)&qrow[8 * half + 32 * ks],
                   *(const v8bf*)&qrow[16 + 8 * half + 32 * ks]);

  v16bf bz;
#pragma unroll
  for (int e = 0; e < 16; ++e) bz[e] = (__bf16)0.0f;

  for (int nt = 0; nt < RPAD / 16; ++nt) {
    int r = nt * 16 + l16;
    v8f acc = (v8f){0.f,0.f,0.f,0.f,0.f,0.f,0.f,0.f};
#pragma unroll
    for (int ks = 0; ks < 2; ++ks) {
      v16bf bfr = bz;
      if (r < NREL) {   // divergent loads only; WMMA itself runs with full EXEC
        const __bf16* rp = &relb[(size_t)r * DEPTH + 32 * ks];
        bfr = cat16(*(const v8bf*)&rp[8 * half], *(const v8bf*)&rp[16 + 8 * half]);
      }
      acc = wmma_bf16(af[ks], bfr, acc);
    }
#pragma unroll
    for (int v = 0; v < 8; ++v) {
      int row = q0 + v + 8 * half;
      int col = nt * 16 + l16;
      if (col < NREL)
        Qrel[((size_t)bh * SEQ + row) * RPAD + col] = (__bf16)acc[v];
    }
  }
}

// ---------------------------------------------------------------- flash attention per (b,h)
// grid (BH, S/64), block 128 = 4 waves; each wave owns 16 q rows.
// K tile staged by TDM (padded rows), V tile staged manually (transposed).
__global__ __launch_bounds__(128)
void attn_kernel(const __bf16* __restrict__ Qb, const __bf16* __restrict__ Kb,
                 const __bf16* __restrict__ Vb, const __bf16* __restrict__ Qrel,
                 __bf16* __restrict__ AO) {
  __shared__ __bf16 Ks[64 * 72];   // [k][d], padded (TDM pad: 32DW+4DW)
  __shared__ __bf16 Vt[64 * 72];   // transposed: [d][k], padded
  __shared__ __bf16 Ps[64 * 72];   // P staging: [wave*16+row][k], padded
  const int tid = threadIdx.x, lane = tid & 31, wave = tid >> 5;
  const int half = lane >> 4, l16 = lane & 15;
  const int bh = blockIdx.x, b = bh >> 4, h = bh & 15;
  const int q0 = blockIdx.y * 64;
  const int qr = q0 + wave * 16 + l16;
  const float inv_scale = 0.125f;  // 1/sqrt(DEPTH)
  const unsigned ksAddr = (unsigned)(size_t)(void*)&Ks[0];

  const __bf16* qrow = &Qb[((size_t)(b * SEQ) + qr) * DM + h * DEPTH];
  v16bf qf[2];
#pragma unroll
  for (int ks = 0; ks < 2; ++ks)
    qf[ks] = cat16(*(const v8bf*)&qrow[8 * half + 32 * ks],
                   *(const v8bf*)&qrow[16 + 8 * half + 32 * ks]);

  v8f o[4];
#pragma unroll
  for (int dt = 0; dt < 4; ++dt) o[dt] = (v8f){0.f,0.f,0.f,0.f,0.f,0.f,0.f,0.f};
  float mrow[8], lrow[8];
#pragma unroll
  for (int v = 0; v < 8; ++v) { mrow[v] = -1e30f; lrow[v] = 0.0f; }

  for (int kb = 0; kb < SEQ; kb += 64) {
    // stage K block (64 rows x 64 bf16, row stride DM) via TDM -> Ks stride 72
    if (wave == 0) {
      tdm_load_2d_bf16(ksAddr, &Kb[((size_t)(b * SEQ) + kb) * DM + h * DEPTH],
                       /*tile_d0=*/64, /*tile_d1=*/64, /*stride0=*/(unsigned long long)DM,
                       /*pad_interval(32DW)=*/4, /*pad_amount(4DW)=*/3);
      __builtin_amdgcn_s_wait_tensorcnt(0);
    }
    // stage V transposed
#pragma unroll
    for (int it = 0; it < 4; ++it) {
      int idx = (tid + it * 128) * 8;
      int kk = idx >> 6, dd = idx & 63;
      size_t g = ((size_t)(b * SEQ) + kb + kk) * DM + h * DEPTH + dd;
      v8bf t = *(const v8bf*)&Vb[g];
#pragma unroll
      for (int e = 0; e < 8; ++e) Vt[(dd + e) * 72 + kk] = t[e];
    }
    __syncthreads();

    // scores S = Q K^T  (16q x 64k per wave)
    v8f s[4];
#pragma unroll
    for (int nt = 0; nt < 4; ++nt) {
      s[nt] = (v8f){0.f,0.f,0.f,0.f,0.f,0.f,0.f,0.f};
#pragma unroll
      for (int ks = 0; ks < 2; ++ks) {
        const __bf16* p = &Ks[(nt * 16 + l16) * 72 + 32 * ks];
        v16bf bfr = cat16(*(const v8bf*)&p[8 * half], *(const v8bf*)&p[16 + 8 * half]);
        s[nt] = wmma_bf16(qf[ks], bfr, s[nt]);
      }
    }

    // add relative-position term (gather from Qrel) and scale
#pragma unroll
    for (int nt = 0; nt < 4; ++nt)
#pragma unroll
      for (int v = 0; v < 8; ++v) {
        int qq = q0 + wave * 16 + v + 8 * half;
        int kk2 = kb + nt * 16 + l16;
        int d = qq - kk2;
        d = d > MAXREL ? MAXREL : (d < -MAXREL ? -MAXREL : d);
        float rv = (float)Qrel[((size_t)bh * SEQ + qq) * RPAD + (d + MAXREL)];
        s[nt][v] = (s[nt][v] + rv) * inv_scale;
      }

    // online softmax (rows live across 16-lane halves -> xor masks 1,2,4,8)
    float alpha[8];
#pragma unroll
    for (int v = 0; v < 8; ++v) {
      float t = fmaxf(fmaxf(s[0][v], s[1][v]), fmaxf(s[2][v], s[3][v]));
      t = fmaxf(t, __shfl_xor(t, 1, 32));
      t = fmaxf(t, __shfl_xor(t, 2, 32));
      t = fmaxf(t, __shfl_xor(t, 4, 32));
      t = fmaxf(t, __shfl_xor(t, 8, 32));
      float mnew = fmaxf(mrow[v], t);
      alpha[v] = __expf(mrow[v] - mnew);
      mrow[v] = mnew;
    }
#pragma unroll
    for (int nt = 0; nt < 4; ++nt)
#pragma unroll
      for (int v = 0; v < 8; ++v)
        s[nt][v] = __expf(s[nt][v] - mrow[v]);
#pragma unroll
    for (int v = 0; v < 8; ++v) {
      float rs = s[0][v] + s[1][v] + s[2][v] + s[3][v];
      rs += __shfl_xor(rs, 1, 32);
      rs += __shfl_xor(rs, 2, 32);
      rs += __shfl_xor(rs, 4, 32);
      rs += __shfl_xor(rs, 8, 32);
      lrow[v] = lrow[v] * alpha[v] + rs;
    }
#pragma unroll
    for (int dt = 0; dt < 4; ++dt)
#pragma unroll
      for (int v = 0; v < 8; ++v) o[dt][v] *= alpha[v];

    // C-layout -> A-layout for P via LDS
#pragma unroll
    for (int nt = 0; nt < 4; ++nt)
#pragma unroll
      for (int v = 0; v < 8; ++v)
        Ps[(wave * 16 + v + 8 * half) * 72 + nt * 16 + l16] = (__bf16)s[nt][v];
    __syncthreads();

    // O += P V
#pragma unroll
    for (int ks = 0; ks < 2; ++ks) {
      const __bf16* pp = &Ps[(wave * 16 + l16) * 72 + 32 * ks];
      v16bf pf = cat16(*(const v8bf*)&pp[8 * half], *(const v8bf*)&pp[16 + 8 * half]);
#pragma unroll
      for (int dt = 0; dt < 4; ++dt) {
        const __bf16* vp = &Vt[(dt * 16 + l16) * 72 + 32 * ks];
        v16bf bfr = cat16(*(const v8bf*)&vp[8 * half], *(const v8bf*)&vp[16 + 8 * half]);
        o[dt] = wmma_bf16(pf, bfr, o[dt]);
      }
    }
    __syncthreads();
  }

  // normalize and store to attn-output buffer [B*S, DM] bf16
#pragma unroll
  for (int dt = 0; dt < 4; ++dt)
#pragma unroll
    for (int v = 0; v < 8; ++v) {
      int qq = q0 + wave * 16 + v + 8 * half;
      int dd = dt * 16 + l16;
      float val = o[dt][v] / lrow[v];
      AO[((size_t)(b * SEQ) + qq) * DM + h * DEPTH + dd] = (__bf16)val;
    }
}

// ---------------------------------------------------------------- launch
extern "C" void kernel_launch(void* const* d_in, const int* in_sizes, int n_in,
                              void* d_out, int out_size, void* d_ws, size_t ws_size,
                              hipStream_t stream) {
  const float* x   = (const float*)d_in[0];
  const float* Wq  = (const float*)d_in[1];
  const float* bq  = (const float*)d_in[2];
  const float* Wk  = (const float*)d_in[3];
  const float* bk  = (const float*)d_in[4];
  const float* Wv  = (const float*)d_in[5];
  const float* bv  = (const float*)d_in[6];
  const float* Wo  = (const float*)d_in[7];
  const float* bo  = (const float*)d_in[8];
  const float* rel = (const float*)d_in[9];

  char* ws = (char*)d_ws;
  size_t off = 0;
  auto alloc = [&](size_t bytes) -> char* {
    char* p = ws + off;
    off += (bytes + 255) & ~(size_t)255;
    return p;
  };
  const size_t MB2 = (size_t)BATCH * SEQ * DM * 2;        // 8 MB per [B*S, DM] bf16
  __bf16* Xb   = (__bf16*)alloc(MB2);
  __bf16* Wqb  = (__bf16*)alloc((size_t)DM * DM * 2);
  __bf16* Wkb  = (__bf16*)alloc((size_t)DM * DM * 2);
  __bf16* Wvb  = (__bf16*)alloc((size_t)DM * DM * 2);
  __bf16* Wob  = (__bf16*)alloc((size_t)DM * DM * 2);
  __bf16* relb = (__bf16*)alloc((size_t)NREL * DEPTH * 2);
  __bf16* Qb   = (__bf16*)alloc(MB2);
  __bf16* Kb   = (__bf16*)alloc(MB2);
  __bf16* Vb   = (__bf16*)alloc(MB2);
  __bf16* Qr   = (__bf16*)alloc((size_t)BATCH * NH * SEQ * RPAD * 2);  // ~35.7 MB
  __bf16* AO   = (__bf16*)alloc(MB2);
  (void)ws_size; (void)in_sizes; (void)n_in; (void)out_size;

  // 1) convert fp32 -> bf16
  auto cvt = [&](const float* src, __bf16* dst, int n) {
    int blocks = (n + 1023) / 1024;
    cvt_f32_bf16<<<blocks, 256, 0, stream>>>(src, dst, n);
  };
  cvt(x,   Xb,  BATCH * SEQ * DM);
  cvt(Wq,  Wqb, DM * DM);
  cvt(Wk,  Wkb, DM * DM);
  cvt(Wv,  Wvb, DM * DM);
  cvt(Wo,  Wob, DM * DM);
  cvt(rel, relb, NREL * DEPTH);

  const int M = BATCH * SEQ;            // 4096
  dim3 gGemm(DM / 128, M / 128);        // (8, 32)

  // 2) Q/K/V projections (bf16 out)
  wmma_gemm_bias<true><<<gGemm, 256, 0, stream>>>(Xb, Wqb, bq, Qb, M, DM, DM);
  wmma_gemm_bias<true><<<gGemm, 256, 0, stream>>>(Xb, Wkb, bk, Kb, M, DM, DM);
  wmma_gemm_bias<true><<<gGemm, 256, 0, stream>>>(Xb, Wvb, bv, Vb, M, DM, DM);

  // 3) Qrel = per-head Q x rel_table^T
  dim3 gQrel(BATCH * NH, SEQ / 128);    // (64, 8)
  qrel_kernel<<<gQrel, 256, 0, stream>>>(Qb, relb, Qr);

  // 4) attention
  dim3 gAttn(BATCH * NH, SEQ / 64);     // (64, 16)
  attn_kernel<<<gAttn, 128, 0, stream>>>(Qb, Kb, Vb, Qr, AO);

  // 5) output projection (fp32 out)
  wmma_gemm_bias<false><<<gGemm, 256, 0, stream>>>(AO, Wob, bo, d_out, M, DM, DM);
}
